// SpatialCrossAttentionEncoder_16956530885207
// MI455X (gfx1250) — matrix operand order repaired and checked by reading, not compiled, not run on hardware
//
#include <hip/hip_runtime.h>
#include <hip/hip_bf16.h>

// ---------------------------------------------------------------------------
// Problem constants (from reference): L=4, D=512, H=8, S=17, B=16, T=256
// ---------------------------------------------------------------------------
#define LC   4
#define DC   512
#define HC   8
#define SC   17
#define BC   16
#define TC   256
#define DFC  2048
#define MTOK (BC * SC * TC)   // 69632 tokens, divisible by 128

typedef _Float16 half16 __attribute__((ext_vector_type(16)));
typedef float    f32x8  __attribute__((ext_vector_type(8)));
typedef int      v4i_gcc __attribute__((vector_size(16)));  // matches builtin

// ---- CDNA5 async global->LDS path (ASYNCcnt), with safe fallback ----------
#if defined(__gfx1250__) && __has_builtin(__builtin_amdgcn_global_load_async_to_lds_b128)
#define HAVE_ASYNC_LDS 1
#else
#define HAVE_ASYNC_LDS 0
#endif

#if HAVE_ASYNC_LDS
#if __has_builtin(__builtin_amdgcn_s_wait_asynccnt)
#define WAIT_ASYNC(n) __builtin_amdgcn_s_wait_asynccnt(n)
#else
#define WAIT_ASYNC(n) asm volatile("s_wait_asynccnt %0" ::"i"(n) : "memory")
#endif
#else
#define WAIT_ASYNC(n) ((void)0)
#endif

__device__ __forceinline__ void copy16_g2lds(void* lds, const void* g) {
#if HAVE_ASYNC_LDS
  // Builtin signature (from hipcc diagnostic): v4i addrspace(1)* src,
  // v4i addrspace(3)* dst, imm int offset, imm int cpol.
  __builtin_amdgcn_global_load_async_to_lds_b128(
      (__attribute__((address_space(1))) v4i_gcc*)g,
      (__attribute__((address_space(3))) v4i_gcc*)lds, 0, 0);
#else
  *(uint4*)lds = *(const uint4*)g;
#endif
}

// ---------------------------------------------------------------------------
// fp32 -> fp16 weight conversion (one-time per launch)
// ---------------------------------------------------------------------------
__global__ __launch_bounds__(256) void f32_to_f16_kernel(
    const float* __restrict__ in, _Float16* __restrict__ out, int n) {
  for (int i = blockIdx.x * blockDim.x + threadIdx.x; i < n;
       i += gridDim.x * blockDim.x)
    out[i] = (_Float16)in[i];
}

// ---------------------------------------------------------------------------
// LayerNorm over D=512, emit fp16 activations. One block (128 thr) per token.
// ---------------------------------------------------------------------------
__global__ __launch_bounds__(128) void ln_f16_kernel(
    const float* __restrict__ x, const float* __restrict__ s,
    const float* __restrict__ b, _Float16* __restrict__ y) {
  __shared__ float r0[128];
  __shared__ float r1[128];
  const int tid = threadIdx.x;
  const size_t n = blockIdx.x;
  const float4 xv = *(const float4*)(x + n * DC + tid * 4);
  float ls = xv.x + xv.y + xv.z + xv.w;
  float lq = xv.x * xv.x + xv.y * xv.y + xv.z * xv.z + xv.w * xv.w;
  r0[tid] = ls;
  r1[tid] = lq;
  __syncthreads();
  for (int off = 64; off > 0; off >>= 1) {
    if (tid < off) {
      r0[tid] += r0[tid + off];
      r1[tid] += r1[tid + off];
    }
    __syncthreads();
  }
  const float mean = r0[0] * (1.0f / DC);
  const float var  = r1[0] * (1.0f / DC) - mean * mean;
  const float rstd = rsqrtf(var + 1e-5f);
  const int d = tid * 4;
  const float4 sv = *(const float4*)(s + d);
  const float4 bv = *(const float4*)(b + d);
  _Float16* yp = y + n * DC + d;
  yp[0] = (_Float16)((xv.x - mean) * rstd * sv.x + bv.x);
  yp[1] = (_Float16)((xv.y - mean) * rstd * sv.y + bv.y);
  yp[2] = (_Float16)((xv.z - mean) * rstd * sv.z + bv.z);
  yp[3] = (_Float16)((xv.w - mean) * rstd * sv.w + bv.w);
}

// ---------------------------------------------------------------------------
// WMMA f16 GEMM: C[M,N] = A[M,K] * W[N,K]^T  (+ fused epilogue)
//   mode 0: outh = f16(C + bias)
//   mode 1: outh = f16(gelu(C + bias))       (exact erf GELU)
//   mode 2: outf += C + bias                  (fp32 residual accumulate)
// Block: 256 threads (8 waves, 2x4), 128x256 tile; wave = 64x64 (4x4 WMMA
// tiles -> 16 WMMAs per 16 fragment ds_load_b128 per K-step).
// K staged 32 at a time into double-buffered LDS via async global->LDS
// (ASYNCcnt), so stage k+1 copies overlap stage k WMMAs.
// ---------------------------------------------------------------------------
#define BM 128
#define BN 256
#define BK 32
#define LDT 40   // padded K-stride in halves (80B, 16B-aligned rows)

__global__ __launch_bounds__(256) void gemm_wmma_f16(
    const _Float16* __restrict__ A, const _Float16* __restrict__ W,
    const float* __restrict__ bias, _Float16* __restrict__ outh,
    float* __restrict__ outf, int N, int K, int mode) {
  __shared__ __attribute__((aligned(16))) _Float16 As[2 * BM * LDT];
  __shared__ __attribute__((aligned(16))) _Float16 Bs[2 * BN * LDT];

  const int tid  = threadIdx.x;
  const int lane = tid & 31;
  const int wave = tid >> 5;
  const int wm   = wave & 1;   // 2 waves along M
  const int wn   = wave >> 1;  // 4 waves along N
  const int m0   = blockIdx.x * BM;
  const int n0   = blockIdx.y * BN;

  const f32x8 zero = {0.f, 0.f, 0.f, 0.f, 0.f, 0.f, 0.f, 0.f};
  f32x8 acc[4][4];
#pragma unroll
  for (int mi = 0; mi < 4; ++mi)
#pragma unroll
    for (int ni = 0; ni < 4; ++ni) acc[mi][ni] = zero;

  // A-fragment lane mapping (ISA 16-bit A 16x32): lanes 0-15 -> K 0-7,16-23;
  // lanes 16-31 -> K 8-15,24-31. Two contiguous 8-half (16B) runs per lane.
  const int kA = (lane >> 4) * 8;
  // B-fragment lane mapping (same as A): lanes 0-15 -> K 0-7,16-23;
  // lanes 16-31 -> K 8-15,24-31. Two contiguous 8-half (16B) runs per lane.
  const int kB = (lane >> 4) * 8;
  const int rl = lane & 15;

  union Frag {
    half16 v;
    uint4  u[2];
  };

  // Stage one 128x32 A tile (512 x 16B) + one 256x32 W tile (1024 x 16B):
  // 6 copy instructions per thread per stage -> ASYNCcnt 6 per wave/stage.
  auto issue = [&](int k0, int stage) {
    _Float16* as = &As[stage * (BM * LDT)];
    _Float16* bs = &Bs[stage * (BN * LDT)];
#pragma unroll
    for (int i = 0; i < 2; ++i) {
      const int idx  = tid + i * 256;
      const int row  = idx >> 2;
      const int quad = (idx & 3) * 8;
      copy16_g2lds(&as[row * LDT + quad],
                   A + (size_t)(m0 + row) * K + k0 + quad);
    }
#pragma unroll
    for (int i = 0; i < 4; ++i) {
      const int idx  = tid + i * 256;
      const int row  = idx >> 2;
      const int quad = (idx & 3) * 8;
      copy16_g2lds(&bs[row * LDT + quad],
                   W + (size_t)(n0 + row) * K + k0 + quad);
    }
  };

  issue(0, 0);
  int stage = 0;
  for (int k0 = 0; k0 < K; k0 += BK, stage ^= 1) {
    const bool more = (k0 + BK) < K;
    if (more) {
      issue(k0 + BK, stage ^ 1);  // prefetch next stage into other buffer
      WAIT_ASYNC(6);              // current stage's 6 copies retired
    } else {
      WAIT_ASYNC(0);
    }
    __syncthreads();

    const _Float16* as = &As[stage * (BM * LDT)];
    const _Float16* bs = &Bs[stage * (BN * LDT)];

    Frag fa[4], fb[4];
#pragma unroll
    for (int mi = 0; mi < 4; ++mi) {
      const int r = wm * 64 + mi * 16 + rl;
      fa[mi].u[0] = *(const uint4*)(&as[r * LDT + kA]);
      fa[mi].u[1] = *(const uint4*)(&as[r * LDT + kA + 16]);
    }
#pragma unroll
    for (int ni = 0; ni < 4; ++ni) {
      const int r = wn * 64 + ni * 16 + rl;
      fb[ni].u[0] = *(const uint4*)(&bs[r * LDT + kB]);
      fb[ni].u[1] = *(const uint4*)(&bs[r * LDT + kB + 16]);
    }

#pragma unroll
    for (int mi = 0; mi < 4; ++mi)
#pragma unroll
      for (int ni = 0; ni < 4; ++ni)
        acc[mi][ni] = __builtin_amdgcn_wmma_f32_16x16x32_f16(
            false, fa[mi].v, false, fb[ni].v, (short)0, acc[mi][ni], false,
            false);
    __syncthreads();  // all waves done reading before buffer is overwritten
  }

  // ---- epilogue: C 16x16 f32 layout: VGPR j -> M=j (lanes 0-15) / M=j+8
  // (lanes 16-31), N = lane&15 ----
  const int mhalf = (lane >> 4) * 8;
#pragma unroll
  for (int mi = 0; mi < 4; ++mi) {
#pragma unroll
    for (int ni = 0; ni < 4; ++ni) {
      const int col = n0 + wn * 64 + ni * 16 + rl;
      const float bc = bias[col];
#pragma unroll
      for (int j = 0; j < 8; ++j) {
        const int row = m0 + wm * 64 + mi * 16 + mhalf + j;
        const float v = acc[mi][ni][j] + bc;
        const size_t off = (size_t)row * N + col;
        if (mode == 0) {
          outh[off] = (_Float16)v;
        } else if (mode == 1) {
          outh[off] = (_Float16)(0.5f * v * (1.0f + erff(v * 0.70710678f)));
        } else {
          outf[off] += v;
        }
      }
    }
  }
}

// ---------------------------------------------------------------------------
// Spatial attention over S=17 per (b, t, head). <1% of FLOPs -> VALU kernel.
// qkv: [MTOK, 3*D] f16 (q | k | v), o: [MTOK, D] f16.
// ---------------------------------------------------------------------------
__global__ __launch_bounds__(256) void attn_kernel(
    const _Float16* __restrict__ qkv, _Float16* __restrict__ o) {
  __shared__ float q[SC * 64];
  __shared__ float k[SC * 64];
  __shared__ float v[SC * 64];
  __shared__ float sc[SC * 18];

  const int bid = blockIdx.x;
  const int h = bid & (HC - 1);
  const int t = (bid >> 3) & (TC - 1);
  const int b = bid >> 11;  // / (H*T)
  const int tid = threadIdx.x;

  for (int idx = tid; idx < SC * 64; idx += 256) {
    const int s = idx >> 6, d = idx & 63;
    const size_t base = ((size_t)(b * SC + s) * TC + t) * (3 * DC) + h * 64 + d;
    q[idx] = (float)qkv[base];
    k[idx] = (float)qkv[base + DC];
    v[idx] = (float)qkv[base + 2 * DC];
  }
  __syncthreads();

  for (int idx = tid; idx < SC * SC; idx += 256) {
    const int s = idx / SC, u = idx % SC;
    float a = 0.f;
    for (int d = 0; d < 64; ++d) a += q[s * 64 + d] * k[u * 64 + d];
    sc[s * 18 + u] = a * 0.125f;  // 1/sqrt(64)
  }
  __syncthreads();

  if (tid < SC) {
    float mx = -1e30f;
    for (int u = 0; u < SC; ++u) mx = fmaxf(mx, sc[tid * 18 + u]);
    float sum = 0.f;
    for (int u = 0; u < SC; ++u) {
      const float e = __expf(sc[tid * 18 + u] - mx);
      sc[tid * 18 + u] = e;
      sum += e;
    }
    const float inv = 1.0f / sum;
    for (int u = 0; u < SC; ++u) sc[tid * 18 + u] *= inv;
  }
  __syncthreads();

  for (int idx = tid; idx < SC * 64; idx += 256) {
    const int s = idx >> 6, d = idx & 63;
    float a = 0.f;
    for (int u = 0; u < SC; ++u) a += sc[s * 18 + u] * v[u * 64 + d];
    const size_t n = (size_t)(b * SC + s) * TC + t;
    o[n * DC + h * 64 + d] = (_Float16)a;
  }
}

// ---------------------------------------------------------------------------
// Host orchestration
// ---------------------------------------------------------------------------
extern "C" void kernel_launch(void* const* d_in, const int* in_sizes, int n_in,
                              void* d_out, int out_size, void* d_ws,
                              size_t ws_size, hipStream_t stream) {
  (void)in_sizes; (void)n_in; (void)out_size; (void)ws_size;

  const float* x_in   = (const float*)d_in[0];
  const float* qkv_w  = (const float*)d_in[1];
  const float* qkv_b  = (const float*)d_in[2];
  const float* out_w  = (const float*)d_in[3];
  const float* out_b  = (const float*)d_in[4];
  const float* ln1_s  = (const float*)d_in[5];
  const float* ln1_b  = (const float*)d_in[6];
  const float* ln2_s  = (const float*)d_in[7];
  const float* ln2_b  = (const float*)d_in[8];
  const float* w1     = (const float*)d_in[9];
  const float* b1     = (const float*)d_in[10];
  const float* w2     = (const float*)d_in[11];
  const float* b2     = (const float*)d_in[12];

  float* x = (float*)d_out;  // running activations, fp32, updated in place

  // workspace layout (256B aligned slices)
  char* ws = (char*)d_ws;
  size_t off = 0;
  auto take = [&](size_t bytes) {
    char* p = ws + off;
    off += (bytes + 255) & ~(size_t)255;
    return p;
  };
  _Float16* Wqkv16 = (_Float16*)take((size_t)LC * 3 * DC * DC * 2);
  _Float16* Wout16 = (_Float16*)take((size_t)LC * DC * DC * 2);
  _Float16* W1_16  = (_Float16*)take((size_t)LC * DFC * DC * 2);
  _Float16* W2_16  = (_Float16*)take((size_t)LC * DC * DFC * 2);
  _Float16* h16    = (_Float16*)take((size_t)MTOK * DC * 2);
  _Float16* o16    = (_Float16*)take((size_t)MTOK * DC * 2);
  _Float16* qkv16  = (_Float16*)take((size_t)MTOK * 3 * DC * 2);
  _Float16* ff16   = (_Float16*)take((size_t)MTOK * DFC * 2);

  // seed x = input (fp32 D2D copy)
  (void)hipMemcpyAsync(x, x_in, (size_t)MTOK * DC * sizeof(float),
                       hipMemcpyDeviceToDevice, stream);

  // one-time weight conversion fp32 -> fp16
  f32_to_f16_kernel<<<2048, 256, 0, stream>>>(qkv_w, Wqkv16, LC * 3 * DC * DC);
  f32_to_f16_kernel<<<2048, 256, 0, stream>>>(out_w, Wout16, LC * DC * DC);
  f32_to_f16_kernel<<<2048, 256, 0, stream>>>(w1, W1_16, LC * DFC * DC);
  f32_to_f16_kernel<<<2048, 256, 0, stream>>>(w2, W2_16, LC * DC * DFC);

  const int MB = MTOK / BM;  // 544 block rows

  for (int l = 0; l < LC; ++l) {
    // ---- pre-LN -> MHA -> residual ----
    ln_f16_kernel<<<MTOK, 128, 0, stream>>>(x, ln1_s + l * DC, ln1_b + l * DC,
                                            h16);
    gemm_wmma_f16<<<dim3(MB, (3 * DC) / BN), 256, 0, stream>>>(
        h16, Wqkv16 + (size_t)l * 3 * DC * DC, qkv_b + l * 3 * DC, qkv16,
        nullptr, 3 * DC, DC, 0);
    attn_kernel<<<BC * TC * HC, 256, 0, stream>>>(qkv16, o16);
    gemm_wmma_f16<<<dim3(MB, DC / BN), 256, 0, stream>>>(
        o16, Wout16 + (size_t)l * DC * DC, out_b + l * DC, nullptr, x, DC, DC,
        2);
    // ---- pre-LN -> GELU FFN -> residual ----
    ln_f16_kernel<<<MTOK, 128, 0, stream>>>(x, ln2_s + l * DC, ln2_b + l * DC,
                                            h16);
    gemm_wmma_f16<<<dim3(MB, DFC / BN), 256, 0, stream>>>(
        h16, W1_16 + (size_t)l * DFC * DC, b1 + l * DFC, ff16, nullptr, DFC, DC,
        1);
    gemm_wmma_f16<<<dim3(MB, DC / BN), 256, 0, stream>>>(
        ff16, W2_16 + (size_t)l * DC * DFC, b2 + l * DC, nullptr, x, DC, DFC,
        2);
  }
}
